// BlockAttention_12979391169283
// MI455X (gfx1250) — matrix operand order repaired and checked by reading, not compiled, not run on hardware
//
#include <hip/hip_runtime.h>
#include <hip/hip_bf16.h>

// ---------------------------------------------------------------------------
// BlockAttention prefill for gfx1250 (MI455X):
//   - bf16 WMMA (v_wmma_f32_16x16x32_bf16) flash-attention over a block-causal
//     staircase mask (32-token blocks => no per-element masking needed).
//   - f32->bf16 conversion while staging K/V through LDS; prefers the
//     hardware packed convert (v_cvt_pk_bf16_f32) when the builtin exists.
//   - V staged transposed so PV B-fragments are contiguous bf16 pairs.
//   - online softmax kept in registers of lanes 0-15, broadcast via __shfl.
//   - cache scatter done by two small float4 kernels (copy then scatter).
// ---------------------------------------------------------------------------

#define NUM_HEADS 16
#define NUM_KV_HEADS 4
#define HEAD_DIM 128
#define SEQ_T 4096
#define BLK_LEN 32
#define QSTRIDE (NUM_HEADS * HEAD_DIM)     // 2048
#define KVSTRIDE (NUM_KV_HEADS * HEAD_DIM) // 512
#define ATTN_SCALE 0.08838834764831845f

typedef __attribute__((ext_vector_type(16))) __bf16 v16bf;
typedef __attribute__((ext_vector_type(8)))  float  v8f;

union Frag {
    unsigned int d[8];  // 8 dwords = 16 packed bf16
    v16bf        b;
};

// f32x2 -> packed bf16x2 (lo = a, hi = b)
#if defined(__gfx1250__) && __has_builtin(__builtin_amdgcn_cvt_pk_bf16_f32)
typedef __attribute__((ext_vector_type(2))) __bf16 v2bf;
__device__ __forceinline__ unsigned int pack2(float a, float b) {
    union { v2bf v; unsigned int u; } cvt;
    cvt.v = __builtin_amdgcn_cvt_pk_bf16_f32(a, b);
    return cvt.u;
}
#else
__device__ __forceinline__ unsigned short f2bf(float f) {
    unsigned int x = __float_as_uint(f);
    unsigned int r = x + 0x7FFFu + ((x >> 16) & 1u);
    return (unsigned short)(r >> 16);
}
__device__ __forceinline__ unsigned int pack2(float a, float b) {
    return (unsigned int)f2bf(a) | ((unsigned int)f2bf(b) << 16);
}
#endif

// A-matrix (16x32 bf16) K-index for a given lane/vgpr, per ISA 7.12.2:
//   lanes 0-15 : VGPR0-3 -> K=0..7,  VGPR4-7 -> K=16..23
//   lanes16-31 : VGPR0-3 -> K=8..15, VGPR4-7 -> K=24..31
__device__ __forceinline__ int apackK(int lane, int vv) {
    int base = (vv < 4) ? (2 * vv) : (16 + 2 * (vv - 4));
    return base + ((lane >> 4) << 3);
}

__global__ void __launch_bounds__(128)
block_attn_kernel(const float* __restrict__ q,
                  const float* __restrict__ k,
                  const float* __restrict__ v,
                  float* __restrict__ out) {
    // Shared staging: K row-major [key][d], V transposed [d][key], per-wave
    // score scratch and bf16 probability tiles.
    __shared__ __align__(16) unsigned short Klds[32 * 128];   // 8 KB bf16
    __shared__ __align__(16) unsigned short Vlds[128 * 32];   // 8 KB bf16 (transposed)
    __shared__ __align__(16) float          Slds[4][16 * 32]; // 8 KB f32
    __shared__ __align__(16) unsigned short Plds[4][16 * 32]; // 4 KB bf16

    const int tid   = threadIdx.x;
    const int w     = tid >> 5;       // wave id 0..3
    const int lane  = tid & 31;
    const int lhalf = lane >> 4;      // 0: lanes 0-15, 1: lanes 16-31
    const int l16   = lane & 15;
    const int qtile = blockIdx.x;     // 64 queries per tile
    const int h     = blockIdx.y;
    const int hkv   = h >> 2;         // GQA group of 4

    // -------- load this wave's 16x128 Q tile as 4 bf16 A-fragments ---------
    const int m    = l16;
    const int qrow = qtile * 64 + w * 16 + m;
    const float* qbase = q + (size_t)qrow * QSTRIDE + h * HEAD_DIM;
    Frag qa[4];
#pragma unroll
    for (int c = 0; c < 4; ++c) {
#pragma unroll
        for (int vv = 0; vv < 8; ++vv) {
            int kk = apackK(lane, vv) + 32 * c;
            float2 f = *reinterpret_cast<const float2*>(qbase + kk);
            qa[c].d[vv] = pack2(f.x, f.y);
        }
    }

    v8f oacc[8];
    v8f zero8 = {0.f, 0.f, 0.f, 0.f, 0.f, 0.f, 0.f, 0.f};
#pragma unroll
    for (int nt = 0; nt < 8; ++nt) oacc[nt] = zero8;

    float mrun = -1e30f;  // valid in lanes 0-15 (one row each)
    float lrun = 0.f;

    const int jmaxw = (qtile * 64 + w * 16) >> 5;  // last allowed key block
    const int jend  = qtile * 2 + 1;               // last block any wave needs

    for (int j = 0; j <= jend; ++j) {
        __syncthreads();  // previous iteration's LDS reads complete

        // ---------------- cooperative K stage (row-major bf16) ------------
        const float* kblk = k + (size_t)(j * 32) * KVSTRIDE + hkv * HEAD_DIM;
#pragma unroll
        for (int i = 0; i < 8; ++i) {
            int idx = i * 512 + tid * 4;           // 0..4095
            int key = idx >> 7, d = idx & 127;
            float4 f = *reinterpret_cast<const float4*>(
                kblk + (size_t)key * KVSTRIDE + d);
            uint2 p;
            p.x = pack2(f.x, f.y);
            p.y = pack2(f.z, f.w);
            *reinterpret_cast<uint2*>(&Klds[idx]) = p;
        }
        // ---------------- cooperative V stage (transposed bf16) -----------
        const float* vblk = v + (size_t)(j * 32) * KVSTRIDE + hkv * HEAD_DIM;
#pragma unroll
        for (int i = 0; i < 16; ++i) {
            int d  = tid;        // 0..127, coalesced across lanes
            int kp = 2 * i;      // key pair
            float a = vblk[(size_t)(kp + 0) * KVSTRIDE + d];
            float b = vblk[(size_t)(kp + 1) * KVSTRIDE + d];
            *reinterpret_cast<unsigned int*>(&Vlds[d * 32 + kp]) = pack2(a, b);
        }
        // prefetch next key block into GL2 while we compute this one
        if (j < jend) {
            __builtin_prefetch(kblk + 32 * KVSTRIDE + tid * 32, 0, 1);
            __builtin_prefetch(vblk + 32 * KVSTRIDE + tid * 32, 0, 1);
        }
        __syncthreads();

        if (j > jmaxw) continue;  // staircase mask at block granularity

        // ---------------- S = Q * K^T (2 N-tiles x 4 K-chunks) ------------
        v8f sacc[2];
#pragma unroll
        for (int nt = 0; nt < 2; ++nt) {
            sacc[nt] = zero8;
            int key = l16 + nt * 16;
#pragma unroll
            for (int c = 0; c < 4; ++c) {
                Frag bf;
#pragma unroll
                for (int vv = 0; vv < 8; ++vv) {
                    int d = apackK(lane, vv) + 32 * c;  // B: same K order as A
                    bf.d[vv] = *reinterpret_cast<const unsigned int*>(
                        &Klds[key * 128 + d]);
                }
                sacc[nt] = __builtin_amdgcn_wmma_f32_16x16x32_bf16(
                    false, qa[c].b, false, bf.b, (short)0, sacc[nt],
                    false, false);
            }
        }

        // spill scaled scores to per-wave scratch (C layout -> row-major)
        float* srow = &Slds[w][0];
#pragma unroll
        for (int nt = 0; nt < 2; ++nt)
#pragma unroll
            for (int r = 0; r < 8; ++r)
                srow[(r + lhalf * 8) * 32 + l16 + nt * 16] =
                    sacc[nt][r] * ATTN_SCALE;

        // ---------------- online softmax: lane = row (lanes 0-15) ---------
        float scaleF = 1.0f;
        if (lhalf == 0) {
            float rowv[32];
            float mx = mrun;
#pragma unroll
            for (int i = 0; i < 32; ++i) {
                rowv[i] = srow[l16 * 32 + i];
                mx = fmaxf(mx, rowv[i]);
            }
            scaleF = __expf(mrun - mx);
            float ls = 0.f;
#pragma unroll
            for (int i = 0; i < 32; ++i) {
                float p = __expf(rowv[i] - mx);
                ls += p;
                rowv[i] = p;
            }
            lrun = lrun * scaleF + ls;
            mrun = mx;
            unsigned short* prow = &Plds[w][l16 * 32];
#pragma unroll
            for (int i = 0; i < 16; ++i)
                *reinterpret_cast<unsigned int*>(&prow[2 * i]) =
                    pack2(rowv[2 * i], rowv[2 * i + 1]);
        }

        // rescale running O by exp(m_old - m_new), row scale via shuffle
#pragma unroll
        for (int r = 0; r < 8; ++r) {
            float sr = __shfl(scaleF, r + lhalf * 8, 32);
#pragma unroll
            for (int nt = 0; nt < 8; ++nt) oacc[nt][r] *= sr;
        }

        // ---------------- O += P * V (1 A-frag, 8 N-tiles) ----------------
        Frag pf;
#pragma unroll
        for (int vv = 0; vv < 8; ++vv)
            pf.d[vv] = *reinterpret_cast<const unsigned int*>(
                &Plds[w][l16 * 32 + apackK(lane, vv)]);
#pragma unroll
        for (int nt = 0; nt < 8; ++nt) {
            Frag vf;
            int d = l16 + nt * 16;
#pragma unroll
            for (int vv = 0; vv < 8; ++vv)
                vf.d[vv] = *reinterpret_cast<const unsigned int*>(
                    &Vlds[d * 32 + apackK(lane, vv)]);
            oacc[nt] = __builtin_amdgcn_wmma_f32_16x16x32_bf16(
                false, pf.b, false, vf.b, (short)0, oacc[nt], false, false);
        }
    }

    // ---------------- epilogue: O /= l, store -----------------------------
    float invl = 1.0f;
    if (lhalf == 0) invl = 1.0f / lrun;
#pragma unroll
    for (int r = 0; r < 8; ++r) {
        float ir  = __shfl(invl, r + lhalf * 8, 32);
        int  orow = qtile * 64 + w * 16 + r + lhalf * 8;
        float* ob = out + (size_t)orow * QSTRIDE + h * HEAD_DIM + l16;
#pragma unroll
        for (int nt = 0; nt < 8; ++nt) ob[nt * 16] = oacc[nt][r] * ir;
    }
}

// ---------------- cache copy + scatter ------------------------------------

__global__ void __launch_bounds__(256)
copy_caches_kernel(const float4* __restrict__ kc, const float4* __restrict__ vc,
                   float4* __restrict__ okc, float4* __restrict__ ovc, int n4) {
    int i = blockIdx.x * blockDim.x + threadIdx.x;
    if (i < n4) {
        okc[i] = kc[i];
        ovc[i] = vc[i];
    }
}

__global__ void __launch_bounds__(256)
scatter_kv_kernel(const float4* __restrict__ k, const float4* __restrict__ v,
                  const int* __restrict__ slot,
                  float4* __restrict__ okc, float4* __restrict__ ovc, int n4) {
    int i = blockIdx.x * blockDim.x + threadIdx.x;
    if (i < n4) {
        int t = i >> 7;          // 128 float4 per row of 512 floats
        int c = i & 127;
        int s = slot[t];
        okc[(size_t)s * 128 + c] = k[i];
        ovc[(size_t)s * 128 + c] = v[i];
    }
}

extern "C" void kernel_launch(void* const* d_in, const int* in_sizes, int n_in,
                              void* d_out, int out_size, void* d_ws, size_t ws_size,
                              hipStream_t stream) {
    (void)in_sizes; (void)n_in; (void)out_size; (void)d_ws; (void)ws_size;

    const float* q  = (const float*)d_in[0];
    const float* k  = (const float*)d_in[1];
    const float* v  = (const float*)d_in[2];
    const float* kc = (const float*)d_in[3];
    const float* vc = (const float*)d_in[4];
    const int* slot = (const int*)d_in[5];

    float* out = (float*)d_out;
    float* o   = out;                                       // 4096*2048
    float* okc = out + (size_t)SEQ_T * QSTRIDE;             // 8192*512
    float* ovc = okc + (size_t)8192 * KVSTRIDE;             // 8192*512

    // 1) copy input caches to output, 2) scatter K/V rows, 3) attention.
    int n4c = (8192 * KVSTRIDE) / 4;  // 1,048,576 float4 per cache
    copy_caches_kernel<<<(n4c + 255) / 256, 256, 0, stream>>>(
        (const float4*)kc, (const float4*)vc, (float4*)okc, (float4*)ovc, n4c);

    int n4s = (SEQ_T * KVSTRIDE) / 4; // 524,288 float4 per tensor
    scatter_kv_kernel<<<(n4s + 255) / 256, 256, 0, stream>>>(
        (const float4*)k, (const float4*)v, slot, (float4*)okc, (float4*)ovc, n4s);

    dim3 grid(SEQ_T / 64, NUM_HEADS, 1);  // 64 x 16 workgroups
    block_attn_kernel<<<grid, 128, 0, stream>>>(q, k, v, o);
}
